// GINEAttentionLayer_56221121904770
// MI455X (gfx1250) — matrix operand order repaired and checked by reading, not compiled, not run on hardware
//
#include <hip/hip_runtime.h>
#include <math.h>
#include <stdint.h>

// Problem constants (match reference)
constexpr int BB = 8;
constexpr int NN = 4096;
constexpr int EE = 65536;
constexpr int HH = 128;
constexpr int NHEADS = 4;
constexpr int HD = 32;          // HH / NHEADS
constexpr int KATTN = 3 * HH;   // 384
constexpr float BN_EPS = 1e-5f;
constexpr int TGRP = 8;         // 64-edge groups per WG in edge_attn
constexpr int RGRP = 4;         // 64-row groups per WG in node_mlp

typedef __attribute__((ext_vector_type(16))) __bf16 v16bf;
typedef __attribute__((ext_vector_type(8)))  __bf16 v8bf;
typedef __attribute__((ext_vector_type(8)))  float  v8f;
typedef __attribute__((ext_vector_type(4)))  unsigned u32x4;
typedef __attribute__((ext_vector_type(8)))  int     i32x8;
typedef __attribute__((ext_vector_type(4)))  int     i32x4;

__device__ __forceinline__ v8f wmma_bf16(v16bf a, v16bf b, v8f c) {
  return __builtin_amdgcn_wmma_f32_16x16x32_bf16(
      false, a, false, b, (short)0, c, false, false);
}

// ---- Tensor Data Mover: 2D bf16 tile Global->LDS (D# per CDNA5 ISA 8.3/8.4)
// 6-arg builtin variant (clang-23 / therock-10.0 headers).
__device__ __forceinline__ void tdm_load_2d_bf16(
    unsigned lds_off, const void* gsrc,
    unsigned tensor_d0, unsigned tensor_d1,
    unsigned tile_d0, unsigned tile_d1, unsigned stride0_elems) {
  unsigned long long ga = (unsigned long long)(size_t)gsrc;
  u32x4 g0;
  g0[0] = 1u;                                                  // count=1, load
  g0[1] = lds_off;                                             // LDS byte addr
  g0[2] = (unsigned)ga;                                        // gaddr[31:0]
  g0[3] = (unsigned)((ga >> 32) & 0x01FFFFFFu) | (2u << 30);   // gaddr[56:32]|type=2
  i32x8 g1;
  g1[0] = 0x00010000;                                          // data_size=2B
  g1[1] = (int)((tensor_d0 & 0xFFFFu) << 16);                  // tensor_dim0 @bit48
  g1[2] = (int)((tensor_d0 >> 16) | ((tensor_d1 & 0xFFFFu) << 16));
  g1[3] = (int)((tensor_d1 >> 16) | ((tile_d0 & 0xFFFFu) << 16));
  g1[4] = (int)(tile_d1 & 0xFFFFu);                            // tile_dim1 (dim2=0)
  g1[5] = (int)stride0_elems;                                  // dim0 stride
  g1[6] = 0;
  g1[7] = 0;
  i32x4 z4 = {0, 0, 0, 0};
  i32x8 z8 = {0, 0, 0, 0, 0, 0, 0, 0};
  __builtin_amdgcn_tensor_load_to_lds(g0, g1, z4, z4, z8, 0);
}

// Monotone order-preserving float<->uint key for atomic max on floats
__device__ __forceinline__ unsigned f2key(float f) {
  unsigned b = __float_as_uint(f);
  return (b & 0x80000000u) ? ~b : (b | 0x80000000u);
}
__device__ __forceinline__ float key2f(unsigned k) {
  unsigned b = (k & 0x80000000u) ? (k ^ 0x80000000u) : ~k;
  return __uint_as_float(b);
}

// ---------------------------------------------------------------------------
// Kernel 0a: init workspace (max keys, softmax sums, aggregation buffer)
// ---------------------------------------------------------------------------
__global__ void init_ws(unsigned* __restrict__ node_max_key,
                        float* __restrict__ node_sum,
                        float* __restrict__ agg) {
  int i = blockIdx.x * blockDim.x + threadIdx.x;
  if (i < BB * NN * NHEADS) {
    node_max_key[i] = 0u;   // below key(-FLT_MAX): identity for atomic max
    node_sum[i] = 0.0f;
  }
  if (i < BB * NN * HH) agg[i] = 0.0f;
}

// ---------------------------------------------------------------------------
// Kernel 0b: one-time fp32 -> bf16 weight conversion into workspace
// ---------------------------------------------------------------------------
__global__ void convert_weights(const float* __restrict__ Wa,
                                const float* __restrict__ W1,
                                const float* __restrict__ W2,
                                __bf16* __restrict__ wb) {
  int i = blockIdx.x * blockDim.x + threadIdx.x;
  if (i < HH * KATTN) wb[i] = (__bf16)Wa[i];
  if (i < HH * HH) {
    wb[HH * KATTN + i] = (__bf16)W1[i];
    wb[HH * KATTN + HH * HH + i] = (__bf16)W2[i];
  }
}

// ---------------------------------------------------------------------------
// Kernel 1: edge attention projection. TDM loads the whole bf16 W_attn into
// LDS once; each wave runs TGRP 16-edge WMMA tiles (K-loop over 384), then
// leaky-relu + head-dot + masked logits + atomic max.
// grid = (EE/(64*TGRP), BB), block = 128 (4 waves)
// dyn-LDS = 128*384*2 (W) + 4*16*128*4 (proj) + 128*4 (a) = 131584 B
// ---------------------------------------------------------------------------
__global__ __launch_bounds__(128) void edge_attn(
    const float* __restrict__ h, const int* __restrict__ edge_index,
    const float* __restrict__ edge_attr, const unsigned char* __restrict__ edge_mask,
    const __bf16* __restrict__ wb_attn, const float* __restrict__ a_vec,
    float* __restrict__ e_buf, unsigned* __restrict__ node_max_key) {
  extern __shared__ char smem[];
  __bf16* Wl   = (__bf16*)smem;                                  // [128][384]
  float*  proj = (float*)(smem + HH * KATTN * 2);                // [4][16][128]
  float*  aS   = (float*)(smem + HH * KATTN * 2 + 4 * 16 * 128 * 4);

  const int lane = threadIdx.x & 31;
  const int wave = threadIdx.x >> 5;
  const int b    = blockIdx.y;
  const int mrow = lane & 15;
  const int g    = lane >> 4;

  if (threadIdx.x == 0) {
    tdm_load_2d_bf16((unsigned)(size_t)Wl, wb_attn, KATTN, HH, KATTN, HH, KATTN);
  }
  aS[threadIdx.x] = a_vec[threadIdx.x];            // 128 == NHEADS*HD
  if (threadIdx.x == 0) __builtin_amdgcn_s_wait_tensorcnt(0);
  __syncthreads();

  float* projW = proj + wave * 16 * 128;

  for (int t = 0; t < TGRP; ++t) {
    const int eb  = ((blockIdx.x * TGRP + t) * 4 + wave) * 16;
    const int eid = eb + mrow;
    const int src = edge_index[b * 2 * EE + eid];
    const int dst = edge_index[b * 2 * EE + EE + eid];
    const float* hD = h + ((size_t)b * NN + dst) * HH;
    const float* hS = h + ((size_t)b * NN + src) * HH;
    const float* ea = edge_attr + ((size_t)b * EE + eid) * HH;
    __builtin_prefetch(ea, 0, 1);
    __builtin_prefetch(ea + 64, 0, 1);

    v8f c[8] = {};
    for (int kb = 0; kb < KATTN / 32; ++kb) {
      // A fragment: one contiguous 16-float run (k = 16*g + s, lane half g)
      int off0 = kb * 32 + (g << 4);
      int off1 = off0 + 8;
      const float* p0 = off0 < HH ? hD + off0
                      : (off0 < 2 * HH ? hS + off0 - HH : ea + off0 - 2 * HH);
      const float* p1 = off1 < HH ? hD + off1
                      : (off1 < 2 * HH ? hS + off1 - HH : ea + off1 - 2 * HH);
      float4 f0 = ((const float4*)p0)[0], f1 = ((const float4*)p0)[1];
      float4 f2 = ((const float4*)p1)[0], f3 = ((const float4*)p1)[1];
      v16bf af = { (__bf16)f0.x, (__bf16)f0.y, (__bf16)f0.z, (__bf16)f0.w,
                   (__bf16)f1.x, (__bf16)f1.y, (__bf16)f1.z, (__bf16)f1.w,
                   (__bf16)f2.x, (__bf16)f2.y, (__bf16)f2.z, (__bf16)f2.w,
                   (__bf16)f3.x, (__bf16)f3.y, (__bf16)f3.z, (__bf16)f3.w };
#pragma unroll
      for (int nt = 0; nt < 8; ++nt) {
        // B fragment: contiguous 32B in n-major LDS weight image
        v16bf bfr = *(const v16bf*)&Wl[(nt * 16 + mrow) * KATTN + kb * 32 + (g << 4)];
        c[nt] = wmma_bf16(af, bfr, c[nt]);
      }
    }

    // Spill D tiles (lane = col, vgpr r = row g*8+r) to per-wave scratch
#pragma unroll
    for (int nt = 0; nt < 8; ++nt)
#pragma unroll
      for (int r = 0; r < 8; ++r)
        projW[(g * 8 + r) * 128 + nt * 16 + mrow] = c[nt][r];
    __syncthreads();   // uniform trip count; orders spill vs reduction

    // leaky_relu(0.2) -> per-head dot with a -> masked logit + atomic max
    for (int p = lane; p < 64; p += 32) {
      int m = p >> 2, hd = p & 3;
      int em_eid = eb + m;
      bool em = edge_mask[(size_t)b * EE + em_eid] != 0;
      float s = 0.0f;
#pragma unroll
      for (int d = 0; d < HD; ++d) {
        float v = projW[m * 128 + hd * HD + d];
        v = (v > 0.0f) ? v : 0.2f * v;
        s += v * aS[hd * HD + d];
      }
      e_buf[((size_t)b * EE + em_eid) * NHEADS + hd] = em ? s : -3.402823466e38f;
      if (em) {
        int dm = edge_index[b * 2 * EE + EE + em_eid];
        atomicMax(&node_max_key[((size_t)b * NN + dm) * NHEADS + hd], f2key(s));
      }
    }
    __syncthreads();
  }
}

// ---------------------------------------------------------------------------
// Kernel 2: exp(e - max) and atomic-add into per-node denominators
// ---------------------------------------------------------------------------
__global__ void edge_softmax(const int* __restrict__ edge_index,
                             const unsigned char* __restrict__ edge_mask,
                             float* __restrict__ e_buf,
                             const unsigned* __restrict__ node_max_key,
                             float* __restrict__ node_sum) {
  size_t gidx = (size_t)blockIdx.x * blockDim.x + threadIdx.x;
  if (gidx >= (size_t)BB * EE * NHEADS) return;
  int hd = (int)(gidx & 3);
  size_t be = gidx >> 2;                       // b*EE + e
  int b = (int)(be >> 16);                     // EE = 65536
  int e = (int)(be & (EE - 1));
  float ex = 0.0f;
  if (edge_mask[be] != 0) {
    int dst = edge_index[b * 2 * EE + EE + e];
    float m = key2f(node_max_key[((size_t)b * NN + dst) * NHEADS + hd]);
    ex = __expf(e_buf[gidx] - m);
    atomicAdd(&node_sum[((size_t)b * NN + dst) * NHEADS + hd], ex);
  }
  e_buf[gidx] = ex;                            // overwrite logits with exp
}

// ---------------------------------------------------------------------------
// Kernel 3: message scatter — one wave per edge; 4 features per lane.
// ---------------------------------------------------------------------------
__global__ __launch_bounds__(256) void edge_aggregate(
    const float* __restrict__ h, const int* __restrict__ edge_index,
    const float* __restrict__ edge_attr, const float* __restrict__ edge_weight,
    const float* __restrict__ e_buf, const float* __restrict__ node_sum,
    float* __restrict__ agg) {
  int lane = threadIdx.x & 31;
  int wave = threadIdx.x >> 5;
  size_t gid = (size_t)blockIdx.x * 8 + wave;  // b*EE + e
  int b = (int)(gid >> 16);
  int e = (int)(gid & (EE - 1));
  int src = edge_index[b * 2 * EE + e];
  int dst = edge_index[b * 2 * EE + EE + e];

  size_t pg = (gid + 32 < (size_t)BB * EE) ? gid + 32 : gid;
  if (lane < 4) __builtin_prefetch(edge_attr + pg * HH + lane * 32, 0, 0);

  float am = 0.0f;
#pragma unroll
  for (int hd = 0; hd < NHEADS; ++hd) {
    float ex  = e_buf[gid * NHEADS + hd];
    float den = fmaxf(node_sum[((size_t)b * NN + dst) * NHEADS + hd], 1e-8f);
    am += ex / den;
  }
  float coef = 0.25f * am * edge_weight[gid];
  if (coef == 0.0f) return;                    // masked edges: exp == 0

  const float4* hs  = (const float4*)(h + ((size_t)b * NN + src) * HH);
  const float4* eap = (const float4*)(edge_attr + gid * (size_t)HH);
  float* ag = agg + ((size_t)b * NN + dst) * HH;
  float4 hv = hs[lane];
  float4 ev = eap[lane];
  atomicAdd(&ag[lane * 4 + 0], fmaxf(hv.x + ev.x, 0.0f) * coef);
  atomicAdd(&ag[lane * 4 + 1], fmaxf(hv.y + ev.y, 0.0f) * coef);
  atomicAdd(&ag[lane * 4 + 2], fmaxf(hv.z + ev.z, 0.0f) * coef);
  atomicAdd(&ag[lane * 4 + 3], fmaxf(hv.w + ev.w, 0.0f) * coef);
}

// ---------------------------------------------------------------------------
// Kernel 4: fused node MLP. TDM loads bf16 W1+W2 into LDS once; each wave
// runs RGRP 16-row tiles: h_new -> GEMM1 -> BN+ReLU -> GEMM2 -> BN -> mask.
// grid = BB*NN/(64*RGRP), block = 128
// dyn-LDS = 2*128*128*2 (W1,W2) + 4*16*128*2 (acts) = 81920 B
// ---------------------------------------------------------------------------
__global__ __launch_bounds__(128) void node_mlp(
    const float* __restrict__ h, const float* __restrict__ agg,
    const float* __restrict__ eps,
    const __bf16* __restrict__ wb1, const __bf16* __restrict__ wb2,
    const float* __restrict__ b1, const float* __restrict__ g1,
    const float* __restrict__ beta1,
    const float* __restrict__ b2, const float* __restrict__ g2,
    const float* __restrict__ beta2,
    const unsigned char* __restrict__ node_mask, float* __restrict__ out) {
  extern __shared__ char smem[];
  __bf16* Wl1 = (__bf16*)smem;                 // [128][128]
  __bf16* Wl2 = Wl1 + HH * HH;                 // [128][128]
  __bf16* xa  = Wl2 + HH * HH;                 // [4][16][128]

  const int lane = threadIdx.x & 31;
  const int wave = threadIdx.x >> 5;
  const int mrow = lane & 15;
  const int g    = lane >> 4;
  const float onePlusEps = 1.0f + eps[0];
  const float bnscale = rsqrtf(1.0f + BN_EPS);

  if (threadIdx.x == 0) {
    tdm_load_2d_bf16((unsigned)(size_t)Wl1, wb1, HH, HH, HH, HH, HH);
    tdm_load_2d_bf16((unsigned)(size_t)Wl2, wb2, HH, HH, HH, HH, HH);
    __builtin_amdgcn_s_wait_tensorcnt(0);
  }
  __syncthreads();

  __bf16* xw = xa + wave * 16 * 128;

  for (int t = 0; t < RGRP; ++t) {
    const int rb = (blockIdx.x * RGRP + t) * 64 + wave * 16;

    // Stage h_new = (1+eps)h + agg as bf16: coalesced b128 loads, b128 stores
#pragma unroll
    for (int it = 0; it < 8; ++it) {
      int flat = it * 256 + lane * 8;
      int row = flat >> 7, col = flat & 127;
      size_t gofs = (size_t)(rb + row) * HH + col;
      float4 u0 = *(const float4*)(h + gofs);
      float4 u1 = *(const float4*)(h + gofs + 4);
      float4 a0 = *(const float4*)(agg + gofs);
      float4 a1 = *(const float4*)(agg + gofs + 4);
      v8bf pk = { (__bf16)(onePlusEps * u0.x + a0.x), (__bf16)(onePlusEps * u0.y + a0.y),
                  (__bf16)(onePlusEps * u0.z + a0.z), (__bf16)(onePlusEps * u0.w + a0.w),
                  (__bf16)(onePlusEps * u1.x + a1.x), (__bf16)(onePlusEps * u1.y + a1.y),
                  (__bf16)(onePlusEps * u1.z + a1.z), (__bf16)(onePlusEps * u1.w + a1.w) };
      *(v8bf*)&xw[row * 128 + col] = pk;
    }

    // ---- GEMM 1: y = x @ W1^T ----
    v8f c1[8] = {};
#pragma unroll
    for (int kb = 0; kb < HH / 32; ++kb) {
      v8bf lo = *(const v8bf*)&xw[mrow * 128 + kb * 32 + (g << 4)];
      v8bf hi = *(const v8bf*)&xw[mrow * 128 + kb * 32 + 8 + (g << 4)];
      v16bf af = __builtin_shufflevector(lo, hi, 0, 1, 2, 3, 4, 5, 6, 7,
                                         8, 9, 10, 11, 12, 13, 14, 15);
#pragma unroll
      for (int nt = 0; nt < 8; ++nt) {
        v16bf bfr = *(const v16bf*)&Wl1[(nt * 16 + mrow) * HH + kb * 32 + (g << 4)];
        c1[nt] = wmma_bf16(af, bfr, c1[nt]);
      }
    }

    // Epilogue 1: +b1, BN, ReLU -> back to LDS as bf16 activations
#pragma unroll
    for (int nt = 0; nt < 8; ++nt)
#pragma unroll
      for (int r = 0; r < 8; ++r) {
        int row = g * 8 + r, col = nt * 16 + mrow;
        float v = c1[nt][r] + b1[col];
        v = v * (g1[col] * bnscale) + beta1[col];
        xw[row * 128 + col] = (__bf16)fmaxf(v, 0.0f);
      }
    __syncthreads();   // uniform; orders epilogue writes vs GEMM2 reads

    // ---- GEMM 2: z = y @ W2^T ----
    v8f c2[8] = {};
#pragma unroll
    for (int kb = 0; kb < HH / 32; ++kb) {
      v8bf lo = *(const v8bf*)&xw[mrow * 128 + kb * 32 + (g << 4)];
      v8bf hi = *(const v8bf*)&xw[mrow * 128 + kb * 32 + 8 + (g << 4)];
      v16bf af = __builtin_shufflevector(lo, hi, 0, 1, 2, 3, 4, 5, 6, 7,
                                         8, 9, 10, 11, 12, 13, 14, 15);
#pragma unroll
      for (int nt = 0; nt < 8; ++nt) {
        v16bf bfr = *(const v16bf*)&Wl2[(nt * 16 + mrow) * HH + kb * 32 + (g << 4)];
        c2[nt] = wmma_bf16(af, bfr, c2[nt]);
      }
    }

    // Epilogue 2: +b2, BN, node mask, store fp32
#pragma unroll
    for (int nt = 0; nt < 8; ++nt)
#pragma unroll
      for (int r = 0; r < 8; ++r) {
        int row = g * 8 + r, col = nt * 16 + mrow;
        int grow = rb + row;
        float v = c2[nt][r] + b2[col];
        v = v * (g2[col] * bnscale) + beta2[col];
        float mk = node_mask[grow] ? 1.0f : 0.0f;
        out[(size_t)grow * HH + col] = v * mk;
      }
    __syncthreads();
  }
}

// ---------------------------------------------------------------------------
extern "C" void kernel_launch(void* const* d_in, const int* in_sizes, int n_in,
                              void* d_out, int out_size, void* d_ws, size_t ws_size,
                              hipStream_t stream) {
  (void)in_sizes; (void)n_in; (void)out_size; (void)ws_size;
  const float*         h           = (const float*)d_in[0];
  const int*           edge_index  = (const int*)d_in[1];
  const float*         edge_attr   = (const float*)d_in[2];
  const unsigned char* node_mask   = (const unsigned char*)d_in[3];
  const unsigned char* edge_mask   = (const unsigned char*)d_in[4];
  const float*         edge_weight = (const float*)d_in[5];
  const float*         eps         = (const float*)d_in[6];
  const float*         W_attn      = (const float*)d_in[7];
  const float*         a_vec       = (const float*)d_in[8];
  const float*         W1          = (const float*)d_in[9];
  const float*         b1          = (const float*)d_in[10];
  const float*         g1          = (const float*)d_in[11];
  const float*         beta1       = (const float*)d_in[12];
  const float*         W2          = (const float*)d_in[13];
  const float*         b2          = (const float*)d_in[14];
  const float*         g2          = (const float*)d_in[15];
  const float*         beta2       = (const float*)d_in[16];
  float* out = (float*)d_out;

  // Workspace carve-up: e_buf | node_max_key | node_sum | agg | bf16 weights
  float*    wsf          = (float*)d_ws;
  float*    e_buf        = wsf;                                   // B*E*NHEADS
  unsigned* node_max_key = (unsigned*)(wsf + (size_t)BB * EE * NHEADS);
  float*    node_sum     = (float*)(node_max_key + (size_t)BB * NN * NHEADS);
  float*    agg          = node_sum + (size_t)BB * NN * NHEADS;   // B*N*H
  __bf16*   wb           = (__bf16*)(agg + (size_t)BB * NN * HH);
  __bf16*   wb_attn      = wb;                                    // [128][384]
  __bf16*   wb1          = wb + (size_t)HH * KATTN;               // [128][128]
  __bf16*   wb2          = wb1 + (size_t)HH * HH;                 // [128][128]

  init_ws<<<(BB * NN * HH + 255) / 256, 256, 0, stream>>>(node_max_key, node_sum, agg);

  convert_weights<<<(HH * KATTN + 255) / 256, 256, 0, stream>>>(W_attn, W1, W2, wb);

  const unsigned smem_attn = HH * KATTN * 2 + 4 * 16 * 128 * 4 + 128 * 4; // 131584
  edge_attn<<<dim3(EE / (64 * TGRP), BB), 128, smem_attn, stream>>>(
      h, edge_index, edge_attr, edge_mask, wb_attn, a_vec, e_buf, node_max_key);

  edge_softmax<<<(BB * EE * NHEADS + 255) / 256, 256, 0, stream>>>(
      edge_index, edge_mask, e_buf, node_max_key, node_sum);

  edge_aggregate<<<(BB * EE) / 8, 256, 0, stream>>>(
      h, edge_index, edge_attr, edge_weight, e_buf, node_sum, agg);

  const unsigned smem_mlp = 2 * HH * HH * 2 + 4 * 16 * 128 * 2;           // 81920
  node_mlp<<<BB * NN / (64 * RGRP), 128, smem_mlp, stream>>>(
      h, agg, eps, wb1, wb2, b1, g1, beta1, b2, g2, beta2, node_mask, out);
}
